// CrossDimensionalGroupedAggregation_89661737271895
// MI455X (gfx1250) — matrix operand order repaired and checked by reading, compiled, not run on hardware
//
#include <hip/hip_runtime.h>
#include <hip/hip_bf16.h>
#include <math.h>

// Sizes from the reference
#define C96   96
#define FGC   192
#define LPIX  4096     // 64*64
#define HW    64
#define NST   16
#define RKC   6
#define KDIR  4
#define DTOT  38       // RK + 2N
#define BATCH 2

typedef __attribute__((ext_vector_type(16))) __bf16 bf16x16;
typedef __attribute__((ext_vector_type(8)))  float  floatx8;

// ---------------------------------------------------------------------------
// WMMA fragment helpers (wave32; contiguous K order k = 16*(lane/16) + s, same in A and B)
// ---------------------------------------------------------------------------

// A: 16x32 bf16. lane%16 = M row, lane/16 selects K-halves.
// VGPR r: base K = (lane/16)*16 + 2r
// Branchless OOB handling: clamp row address (always valid), scale by 0/1 mask.
__device__ __forceinline__ void loadA(bf16x16& a, const float* W, int ldw,
                                      int m0, int k0, int lane, int mMax) {
  int m    = m0 + (lane & 15);
  int half = lane >> 4;
  float msk = (m < mMax) ? 1.0f : 0.0f;
  int mc = (m < mMax) ? m : (mMax - 1);
  const float* row = W + (size_t)mc * ldw;
#pragma unroll
  for (int r = 0; r < 8; ++r) {
    int kb = half * 16 + 2 * r + k0;
    float2 v = *(const float2*)(row + kb);   // 8B-aligned: kb even, ldw multiple of 2
    a[2 * r]     = (__bf16)(v.x * msk);
    a[2 * r + 1] = (__bf16)(v.y * msk);
  }
}

// B: 32x16 bf16, row-major source B[k][n] = src[k*ldb + n].
// lane%16 = N col, VGPR r holds K = (lane/16)*16 + 2r, +1
__device__ __forceinline__ void loadB_rm(bf16x16& bf, const float* B, int ldb,
                                         int k0, int n0, int lane) {
  int n    = n0 + (lane & 15);
  int half = lane >> 4;
#pragma unroll
  for (int r = 0; r < 8; ++r) {
    int k = k0 + half * 16 + 2 * r;
    bf[2 * r]     = (__bf16)B[(size_t)k * ldb + n];
    bf[2 * r + 1] = (__bf16)B[(size_t)(k + 1) * ldb + n];
  }
}

#define WMMA_BF16(acc, a, b) \
  __builtin_amdgcn_wmma_f32_16x16x32_bf16(false, (a), false, (b), (short)0, (acc), false, false)

// Scan-direction l -> pixel bijection (derived from the 4 xs streams)
__device__ __forceinline__ int dirPix(int k, int l) {
  switch (k) {
    case 0:  return l;
    case 1:  return (l & 63) * HW + (l >> 6);
    case 2:  return (LPIX - 1) - l;
    default: { int lr = (LPIX - 1) - l; return (lr & 63) * HW + (lr >> 6); }
  }
}

__device__ __forceinline__ float bnApply(float x, const float* p, int c, int Cn) {
  float g = p[c], bb = p[Cn + c], m = p[2 * Cn + c], v = p[3 * Cn + c];
  return (x - m) * (g * rsqrtf(v + 1e-5f)) + bb;
}

__device__ __forceinline__ float dwc(const float* plane, const float* w,
                                     int ks, int dil, int h, int wp) {
  int half = (ks - 1) >> 1;
  float acc = 0.0f;
  for (int i = 0; i < ks; ++i) {
    int hi = h + (i - half) * dil;
    if (hi < 0 || hi >= HW) continue;
    for (int j = 0; j < ks; ++j) {
      int wi = wp + (j - half) * dil;
      if (wi < 0 || wi >= HW) continue;
      acc += plane[hi * HW + wi] * w[i * ks + j];
    }
  }
  return acc;
}

__device__ __forceinline__ float siluf(float x) { return x / (1.0f + expf(-x)); }

// ---------------------------------------------------------------------------
// K1: y = wg_w @ g + wx_w @ x + wg_b + wx_b          (WMMA GEMM, per batch)
// waves = B(2) * Mt(6) * Nt(256) = 3072
// ---------------------------------------------------------------------------
__global__ void k1_y(const float* __restrict__ g, const float* __restrict__ x,
                     const float* __restrict__ wg_w, const float* __restrict__ wg_b,
                     const float* __restrict__ wx_w, const float* __restrict__ wx_b,
                     float* __restrict__ y) {
  int lane  = threadIdx.x & 31;
  int waveG = blockIdx.x * 8 + (threadIdx.x >> 5);
  int b  = waveG / (6 * 256);
  int r  = waveG % (6 * 256);
  int m0 = (r / 256) * 16;
  int n0 = (r % 256) * 16;
  const float* gB = g + (size_t)b * FGC * LPIX;
  const float* xB = x + (size_t)b * C96 * LPIX;
  floatx8 acc = {};
  bf16x16 a, bb;
#pragma unroll
  for (int kk = 0; kk < FGC; kk += 32) {
    loadA(a, wg_w, FGC, m0, kk, lane, C96);
    loadB_rm(bb, gB, LPIX, kk, n0, lane);
    acc = WMMA_BF16(acc, a, bb);
  }
#pragma unroll
  for (int kk = 0; kk < C96; kk += 32) {
    loadA(a, wx_w, C96, m0, kk, lane, C96);
    loadB_rm(bb, xB, LPIX, kk, n0, lane);
    acc = WMMA_BF16(acc, a, bb);
  }
  float* yB = y + (size_t)b * C96 * LPIX;
  int n = n0 + (lane & 15), half = lane >> 4;
#pragma unroll
  for (int rr = 0; rr < 8; ++rr) {
    int m = m0 + rr + half * 8;
    yB[(size_t)m * LPIX + n] = acc[rr] + wg_b[m] + wx_b[m];
  }
}

// ---------------------------------------------------------------------------
// K2: dr = bn(dw13(y)) + sum_i bn(dw_i(y, dil_i))
// ---------------------------------------------------------------------------
__global__ void k_dr(const float* __restrict__ y,
                     const float* __restrict__ lk_w, const float* __restrict__ lk_bn,
                     const float* __restrict__ br0, const float* __restrict__ br1,
                     const float* __restrict__ br2, const float* __restrict__ br3,
                     const float* __restrict__ br4, const float* __restrict__ br_bn,
                     float* __restrict__ dr) {
  int t = blockIdx.x * blockDim.x + threadIdx.x;
  if (t >= BATCH * C96 * LPIX) return;
  int l = t % LPIX, c = (t / LPIX) % C96, b = t / (LPIX * C96);
  int h = l >> 6, w = l & 63;
  const float* plane = y + ((size_t)b * C96 + c) * LPIX;
  float out = bnApply(dwc(plane, lk_w + (size_t)c * 169, 13, 1, h, w), lk_bn, c, C96);
  const float* ws_[5] = {br0, br1, br2, br3, br4};
  const int ks_[5]  = {5, 7, 3, 3, 3};
  const int dil_[5] = {1, 2, 3, 4, 5};
  for (int i = 0; i < 5; ++i) {
    float acc = dwc(plane, ws_[i] + (size_t)c * ks_[i] * ks_[i], ks_[i], dil_[i], h, w);
    out += bnApply(acc, br_bn + (size_t)i * 4 * C96, c, C96);
  }
  dr[t] = out;
}

// ---------------------------------------------------------------------------
// K3a: xz = in_proj_w(192x96) @ dr      waves = B*12*256 = 6144
// ---------------------------------------------------------------------------
__global__ void k_xz(const float* __restrict__ dr, const float* __restrict__ in_proj_w,
                     float* __restrict__ xz) {
  int lane  = threadIdx.x & 31;
  int waveG = blockIdx.x * 8 + (threadIdx.x >> 5);
  int nt = waveG % 256; int tmp = waveG / 256;
  int mt = tmp % 12;    int b   = tmp / 12;
  int m0 = mt * 16, n0 = nt * 16;
  const float* drB = dr + (size_t)b * C96 * LPIX;
  floatx8 acc = {};
  bf16x16 a, bb;
#pragma unroll
  for (int kk = 0; kk < C96; kk += 32) {
    loadA(a, in_proj_w, C96, m0, kk, lane, 2 * C96);
    loadB_rm(bb, drB, LPIX, kk, n0, lane);
    acc = WMMA_BF16(acc, a, bb);
  }
  float* out = xz + (size_t)b * 2 * C96 * LPIX;
  int n = n0 + (lane & 15), half = lane >> 4;
#pragma unroll
  for (int rr = 0; rr < 8; ++rr) {
    int m = m0 + rr + half * 8;
    out[(size_t)m * LPIX + n] = acc[rr];
  }
}

// ---------------------------------------------------------------------------
// K3b: xc = silu(dw3x3(xp) + dw_b)      (xp = xz channels [0,96))
// ---------------------------------------------------------------------------
__global__ void k_xc(const float* __restrict__ xz, const float* __restrict__ dw_w,
                     const float* __restrict__ dw_b, float* __restrict__ xc) {
  int t = blockIdx.x * blockDim.x + threadIdx.x;
  if (t >= BATCH * C96 * LPIX) return;
  int l = t % LPIX, c = (t / LPIX) % C96, b = t / (LPIX * C96);
  const float* plane = xz + ((size_t)b * 2 * C96 + c) * LPIX;
  float acc = dwc(plane, dw_w + (size_t)c * 9, 3, 1, l >> 6, l & 63) + dw_b[c];
  xc[t] = siluf(acc);
}

// ---------------------------------------------------------------------------
// K3d: dbl[b,k,d,l] = sum_c xproj_w[k,d,c] * xs[b,k,c,l]  (xs gathered from xc)
// waves = B(2)*K(4)*Mt(3)*Nt(256) = 6144 ; M=38 (padded rows -> 0)
// ---------------------------------------------------------------------------
__global__ void k_dbl(const float* __restrict__ xc, const float* __restrict__ xproj_w,
                      float* __restrict__ dbl) {
  int lane  = threadIdx.x & 31;
  int waveG = blockIdx.x * 8 + (threadIdx.x >> 5);
  int nt = waveG % 256; int tmp = waveG / 256;
  int mt = tmp % 3;     tmp /= 3;
  int k  = tmp % KDIR;  int b = tmp / KDIR;
  int m0 = mt * 16, n0 = nt * 16;
  const float* W   = xproj_w + (size_t)k * DTOT * C96;
  const float* xcB = xc + (size_t)b * C96 * LPIX;
  int n = n0 + (lane & 15), half = lane >> 4;
  int pix = dirPix(k, n);                 // n is the absolute scan position l
  floatx8 acc = {};
  bf16x16 a, bb;
#pragma unroll
  for (int c0 = 0; c0 < C96; c0 += 32) {
    loadA(a, W, C96, m0, c0, lane, DTOT);
#pragma unroll
    for (int rr = 0; rr < 8; ++rr) {
      int cc = c0 + half * 16 + 2 * rr;
      bb[2 * rr]     = (__bf16)xcB[(size_t)cc * LPIX + pix];
      bb[2 * rr + 1] = (__bf16)xcB[(size_t)(cc + 1) * LPIX + pix];
    }
    acc = WMMA_BF16(acc, a, bb);
  }
  float* out = dbl + (size_t)(b * KDIR + k) * DTOT * LPIX;
#pragma unroll
  for (int rr = 0; rr < 8; ++rr) {
    int m = m0 + rr + half * 8;
    if (m < DTOT) out[(size_t)m * LPIX + n] = acc[rr];
  }
}

// ---------------------------------------------------------------------------
// K3e: delta = softplus(dtproj_w @ dts + dtproj_b)
// ---------------------------------------------------------------------------
__global__ void k_delta(const float* __restrict__ dbl, const float* __restrict__ dtproj_w,
                        const float* __restrict__ dtproj_b, float* __restrict__ delta) {
  int t = blockIdx.x * blockDim.x + threadIdx.x;
  if (t >= BATCH * KDIR * C96 * LPIX) return;
  int l = t % LPIX;
  int c = (t / LPIX) % C96;
  int k = (t / (LPIX * C96)) % KDIR;
  int b = t / (LPIX * C96 * KDIR);
  const float* dts = dbl + (size_t)(b * KDIR + k) * DTOT * LPIX + l;
  const float* wv  = dtproj_w + ((size_t)(k * C96) + c) * RKC;
  float s = dtproj_b[k * C96 + c];
#pragma unroll
  for (int r = 0; r < RKC; ++r) s += wv[r] * dts[(size_t)r * LPIX];
  delta[t] = (s > 20.0f) ? s : log1pf(expf(s));
}

__global__ void k_zero(float* __restrict__ p, int n) {
  int t = blockIdx.x * blockDim.x + threadIdx.x;
  if (t < n) p[t] = 0.0f;
}

// ---------------------------------------------------------------------------
// K3f: selective scan. 768 chains (b,k,c), 16 lanes per chain (one state n each).
// Cross-lane h*C reduction via shfl_xor; direction streams merged by atomicAdd
// into pixel space (dirPix is a bijection per k).
// ---------------------------------------------------------------------------
__global__ void k_scan(const float* __restrict__ xc, const float* __restrict__ dbl,
                       const float* __restrict__ delta, const float* __restrict__ A_log,
                       const float* __restrict__ Ds, float* __restrict__ ysum) {
  int t = blockIdx.x * blockDim.x + threadIdx.x;   // 12288 threads
  int n = t & 15;
  int chain = t >> 4;                              // 768 chains
  int c = chain % C96;
  int k = (chain / C96) % KDIR;
  int b = chain / (C96 * KDIR);
  float An = -expf(A_log[((size_t)(k * C96) + c) * NST + n]);
  float Dc = Ds[k * C96 + c];
  const float* xcc  = xc + ((size_t)b * C96 + c) * LPIX;
  const float* dlp  = delta + ((size_t)(b * KDIR + k) * C96 + c) * LPIX;
  const float* bsp  = dbl + ((size_t)(b * KDIR + k) * DTOT + RKC + n) * LPIX;
  const float* csp  = bsp + (size_t)NST * LPIX;
  float* ysb = ysum + ((size_t)b * C96 + c) * LPIX;
  float h = 0.0f;
  for (int l = 0; l < LPIX; ++l) {
    float dl = dlp[l];
    int pix  = dirPix(k, l);
    float u  = xcc[pix];
    h = expf(dl * An) * h + (dl * u) * bsp[l];
    float part = h * csp[l];
    part += __shfl_xor(part, 1, 16);
    part += __shfl_xor(part, 2, 16);
    part += __shfl_xor(part, 4, 16);
    part += __shfl_xor(part, 8, 16);
    if (n == 0) atomicAdd(&ysb[pix], part + Dc * u);
  }
}

// ---------------------------------------------------------------------------
// K3g: channel layernorm + silu(z) gate;  yln stored [b][l][c]
// ---------------------------------------------------------------------------
__global__ void k_ln(const float* __restrict__ ysum, const float* __restrict__ xz,
                     const float* __restrict__ ln_g, const float* __restrict__ ln_b,
                     float* __restrict__ yln) {
  int t = blockIdx.x * blockDim.x + threadIdx.x;
  if (t >= BATCH * LPIX) return;
  int b = t / LPIX, l = t % LPIX;
  const float* ys = ysum + (size_t)b * C96 * LPIX + l;
  float s = 0.0f, sq = 0.0f;
  for (int c = 0; c < C96; ++c) { float v = ys[(size_t)c * LPIX]; s += v; sq += v * v; }
  float mu  = s / (float)C96;
  float var = sq / (float)C96 - mu * mu;
  float inv = rsqrtf(var + 1e-5f);
  const float* zc = xz + ((size_t)b * 2 * C96 + C96) * LPIX + l;
  float* o = yln + ((size_t)b * LPIX + l) * C96;
  for (int c = 0; c < C96; ++c) {
    float v = (ys[(size_t)c * LPIX] - mu) * inv * ln_g[c] + ln_b[c];
    o[c] = v * siluf(zc[(size_t)c * LPIX]);
  }
}

// ---------------------------------------------------------------------------
// K3h: ssout = out_proj_w(96x96) @ yln^T    waves = B*6*256 = 3072
// ---------------------------------------------------------------------------
__global__ void k_outproj(const float* __restrict__ yln, const float* __restrict__ out_proj_w,
                          float* __restrict__ ssout) {
  int lane  = threadIdx.x & 31;
  int waveG = blockIdx.x * 8 + (threadIdx.x >> 5);
  int nt = waveG % 256; int tmp = waveG / 256;
  int mt = tmp % 6;     int b   = tmp / 6;
  int m0 = mt * 16, n0 = nt * 16;
  const float* ylnB = yln + (size_t)b * LPIX * C96;
  int n = n0 + (lane & 15), half = lane >> 4;
  const float* rowp = ylnB + (size_t)n * C96;  // channel-contiguous per pixel
  floatx8 acc = {};
  bf16x16 a, bb;
#pragma unroll
  for (int c0 = 0; c0 < C96; c0 += 32) {
    loadA(a, out_proj_w, C96, m0, c0, lane, C96);
#pragma unroll
    for (int rr = 0; rr < 8; ++rr) {
      int cc = c0 + half * 16 + 2 * rr;
      float2 v = *(const float2*)(rowp + cc);   // cc even, rows 96-float aligned
      bb[2 * rr]     = (__bf16)v.x;
      bb[2 * rr + 1] = (__bf16)v.y;
    }
    acc = WMMA_BF16(acc, a, bb);
  }
  float* out = ssout + (size_t)b * C96 * LPIX;
#pragma unroll
  for (int rr = 0; rr < 8; ++rr) {
    int m = m0 + rr + half * 8;
    out[(size_t)m * LPIX + n] = acc[rr];
  }
}

// ---------------------------------------------------------------------------
// K4a: p = sigmoid(bn1(psi_w . relu(y) + psi_b))
// ---------------------------------------------------------------------------
__global__ void k_p(const float* __restrict__ y, const float* __restrict__ psi_w,
                    const float* __restrict__ psi_b, const float* __restrict__ psi_bn,
                    float* __restrict__ p) {
  int t = blockIdx.x * blockDim.x + threadIdx.x;
  if (t >= BATCH * LPIX) return;
  int b = t / LPIX, l = t % LPIX;
  const float* yB = y + (size_t)b * C96 * LPIX + l;
  float s = psi_b[0];
  for (int c = 0; c < C96; ++c) s += psi_w[c] * fmaxf(yB[(size_t)c * LPIX], 0.0f);
  float g = psi_bn[0], bb = psi_bn[1], m = psi_bn[2], v = psi_bn[3];
  float bn = (s - m) * (g * rsqrtf(v + 1e-5f)) + bb;
  p[t] = 1.0f / (1.0f + expf(-bn));
}

// ---------------------------------------------------------------------------
// K4b: out = p * x + relu(ssout)
// ---------------------------------------------------------------------------
__global__ void k_final(const float* __restrict__ p, const float* __restrict__ x,
                        const float* __restrict__ ssout, float* __restrict__ out) {
  int t = blockIdx.x * blockDim.x + threadIdx.x;
  if (t >= BATCH * C96 * LPIX) return;
  int l = t % LPIX;
  int b = t / (LPIX * C96);
  out[t] = p[b * LPIX + l] * x[t] + fmaxf(ssout[t], 0.0f);
}

// ---------------------------------------------------------------------------
extern "C" void kernel_launch(void* const* d_in, const int* in_sizes, int n_in,
                              void* d_out, int out_size, void* d_ws, size_t ws_size,
                              hipStream_t stream) {
  (void)in_sizes; (void)n_in; (void)out_size; (void)ws_size;
  const float* g        = (const float*)d_in[0];
  const float* x        = (const float*)d_in[1];
  const float* wg_w     = (const float*)d_in[2];
  const float* wg_b     = (const float*)d_in[3];
  const float* wx_w     = (const float*)d_in[4];
  const float* wx_b     = (const float*)d_in[5];
  const float* psi_w    = (const float*)d_in[6];
  const float* psi_b    = (const float*)d_in[7];
  const float* psi_bn   = (const float*)d_in[8];
  const float* lk_w     = (const float*)d_in[9];
  const float* lk_bn    = (const float*)d_in[10];
  const float* br0_w    = (const float*)d_in[11];
  const float* br1_w    = (const float*)d_in[12];
  const float* br2_w    = (const float*)d_in[13];
  const float* br3_w    = (const float*)d_in[14];
  const float* br4_w    = (const float*)d_in[15];
  const float* br_bn    = (const float*)d_in[16];
  const float* in_proj_w= (const float*)d_in[17];
  const float* dw_w     = (const float*)d_in[18];
  const float* dw_b     = (const float*)d_in[19];
  const float* xproj_w  = (const float*)d_in[20];
  const float* dtproj_w = (const float*)d_in[21];
  const float* dtproj_b = (const float*)d_in[22];
  const float* A_log    = (const float*)d_in[23];
  const float* Ds       = (const float*)d_in[24];
  const float* ln_g     = (const float*)d_in[25];
  const float* ln_b     = (const float*)d_in[26];
  const float* out_proj_w=(const float*)d_in[27];
  float* out = (float*)d_out;

  // Workspace layout (floats)
  float* ws = (float*)d_ws;
  const size_t NY = (size_t)BATCH * C96 * LPIX;              // 786432
  float* y     = ws;                  // NY
  float* dr    = y     + NY;          // NY
  float* xz    = dr    + NY;          // 2*NY
  float* xc    = xz    + 2 * NY;      // NY
  float* dbl   = xc    + NY;          // B*K*38*L = 1245184
  float* delta = dbl   + (size_t)BATCH * KDIR * DTOT * LPIX; // B*K*C*L
  float* ysum  = delta + (size_t)BATCH * KDIR * C96 * LPIX;  // NY
  float* yln   = ysum  + NY;          // NY
  float* ssout = yln   + NY;          // NY
  float* pbuf  = ssout + NY;          // B*L

  const int TB = 256;
  k1_y<<<384, TB, 0, stream>>>(g, x, wg_w, wg_b, wx_w, wx_b, y);
  k_dr<<<(int)((NY + TB - 1) / TB), TB, 0, stream>>>(y, lk_w, lk_bn, br0_w, br1_w,
                                                     br2_w, br3_w, br4_w, br_bn, dr);
  k_xz<<<768, TB, 0, stream>>>(dr, in_proj_w, xz);
  k_xc<<<(int)((NY + TB - 1) / TB), TB, 0, stream>>>(xz, dw_w, dw_b, xc);
  k_dbl<<<768, TB, 0, stream>>>(xc, xproj_w, dbl);
  k_delta<<<(int)(((size_t)BATCH * KDIR * C96 * LPIX + TB - 1) / TB), TB, 0, stream>>>(
      dbl, dtproj_w, dtproj_b, delta);
  k_zero<<<(int)((NY + TB - 1) / TB), TB, 0, stream>>>(ysum, (int)NY);
  k_scan<<<48, TB, 0, stream>>>(xc, dbl, delta, A_log, Ds, ysum);
  k_ln<<<(BATCH * LPIX + TB - 1) / TB, TB, 0, stream>>>(ysum, xz, ln_g, ln_b, yln);
  k_outproj<<<384, TB, 0, stream>>>(yln, out_proj_w, ssout);
  k_p<<<(BATCH * LPIX + TB - 1) / TB, TB, 0, stream>>>(y, psi_w, psi_b, psi_bn, pbuf);
  k_final<<<(int)((NY + TB - 1) / TB), TB, 0, stream>>>(pbuf, x, ssout, out);
}
